// Attention_66391604462458
// MI455X (gfx1250) — matrix operand order repaired and checked by reading, not compiled, not run on hardware
//
#include <hip/hip_runtime.h>

// CDNA5 (gfx1250) fused axial-attention.
// One workgroup (256 threads = 8 wave32) per (batch b, grid-line i):
//   - LayerNorm + pos for the row-window and column-window token sets (LDS, f16)
//   - per head h: QKV via v_wmma_f32_16x16x32_f16 against f16 K-major weights,
//     64x64 attention (parallel softmax, f32 scores in LDS), +pos, head
//     re-projection, then both cross-attentions; scattered to f32 staging.
//   - finalize kernel: out = x + o1 + o2  (float4 vectorized)

typedef _Float16 h16;
typedef __attribute__((ext_vector_type(16))) _Float16 v16h;
typedef __attribute__((ext_vector_type(8)))  _Float16 v8h;
typedef __attribute__((ext_vector_type(8)))  float    v8f;

#define AST 520          // LN'd activation row stride (halves), 64 x 512 (+8 pad)
#define TST 72           // 64x64 tile buffer row stride (halves)
#define SST 68           // f32 score row stride (floats)
#define TBUF (64 * TST)  // halves per 64x64 tile buffer

// ---- WMMA fragment helpers -------------------------------------------------

__device__ __forceinline__ v16h cat16(v8h lo, v8h hi) {
  return __builtin_shufflevector(lo, hi, 0, 1, 2, 3, 4, 5, 6, 7,
                                         8, 9, 10, 11, 12, 13, 14, 15);
}

// A fragment (16x32 f16, M x K). ISA layout: lane<16 -> M=lane, K {0..7,16..23};
// lane>=16 -> M=lane-16, K {8..15,24..31}.  Two 16B ds loads.
__device__ __forceinline__ v16h load_a_lds(const h16* base, int stride,
                                           int m0, int k0, int lane) {
  int row = m0 + (lane & 15);
  int kk  = k0 + ((lane >> 4) << 3);
  const h16* p = base + row * stride + kk;
  v8h lo = *(const v8h*)p;
  v8h hi = *(const v8h*)(p + 16);
  return cat16(lo, hi);
}

// B fragment (32x16 f16, K x N). N = n0 + lane%16; lane<16 -> K {0..7,16..23},
// lane>=16 -> K {8..15,24..31} (same K order as A).
__device__ __forceinline__ v16h load_b_lds(const h16* base, int stride,
                                           int k0, int n0, int lane) {
  const h16* p = base + n0 + (lane & 15);
  v16h r;
  for (int s = 0; s < 16; ++s)
    r[s] = p[(k0 + ((lane >> 4) << 3) + (s & 7) + ((s >> 3) << 4)) * stride];
  return r;
}

__device__ __forceinline__ v16h load_b_glb(const h16* base, int ld,
                                           int k0, int n0, int lane) {
  const h16* p = base + n0 + (lane & 15);
  v16h r;
  for (int s = 0; s < 16; ++s)
    r[s] = p[(k0 + ((lane >> 4) << 3) + (s & 7) + ((s >> 3) << 4)) * ld];
  return r;
}

__device__ __forceinline__ v8f wmma_f16(v16h a, v16h b, v8f c) {
  return __builtin_amdgcn_wmma_f32_16x16x32_f16(false, a, false, b,
                                                (short)0, c, false, false);
}

// ---- workgroup-level GEMMs (M=64, N=64 output, 16 tiles over 8 waves) ------

// C(64x64) = A_lds(64xK) @ Bglob(Kx ld f16 K-major, cols [bcol, bcol+64))
// f16 result into LDS tile buffer (optionally transposed).
__device__ __forceinline__ void gemm_qkv(const h16* A, int astr,
                                         const h16* Bg, int ldb, int bcol,
                                         int K, h16* out, bool transpose,
                                         int tid) {
  int lane = tid & 31, wave = tid >> 5;
  int e = lane & 15, mo = (lane >> 4) << 3;
  for (int t = wave; t < 16; t += 8) {
    int m0 = (t >> 2) << 4, n0 = (t & 3) << 4;
    v8f c = {0.f, 0.f, 0.f, 0.f, 0.f, 0.f, 0.f, 0.f};
    for (int k0 = 0; k0 < K; k0 += 32) {
      v16h a = load_a_lds(A, astr, m0, k0, lane);
      v16h b = load_b_glb(Bg, ldb, k0, bcol + n0, lane);
      c = wmma_f16(a, b, c);
    }
    for (int r = 0; r < 8; ++r) {
      int m = m0 + r + mo, n = n0 + e;
      if (transpose) out[n * TST + m] = (h16)c[r];
      else           out[m * TST + n] = (h16)c[r];
    }
  }
}

// S(64x64 f32) = Q(64x64) @ KT(64x64 as KxN), K=64, both LDS.
__device__ __forceinline__ void gemm_scores(const h16* Aq, const h16* BkT,
                                            float* Sf, int tid) {
  int lane = tid & 31, wave = tid >> 5;
  int e = lane & 15, mo = (lane >> 4) << 3;
  for (int t = wave; t < 16; t += 8) {
    int m0 = (t >> 2) << 4, n0 = (t & 3) << 4;
    v8f c = {0.f, 0.f, 0.f, 0.f, 0.f, 0.f, 0.f, 0.f};
    for (int k0 = 0; k0 < 64; k0 += 32) {
      v16h a = load_a_lds(Aq, TST, m0, k0, lane);
      v16h b = load_b_lds(BkT, TST, k0, n0, lane);
      c = wmma_f16(a, b, c);
    }
    for (int r = 0; r < 8; ++r)
      Sf[(m0 + r + mo) * SST + n0 + e] = c[r];
  }
}

// X2(64x64 f16) = P(64x64) @ V(64x64) + pos (global f32, 64x64)
__device__ __forceinline__ void gemm_pv_x2(const h16* P, const h16* V,
                                           const float* pos, h16* out, int tid) {
  int lane = tid & 31, wave = tid >> 5;
  int e = lane & 15, mo = (lane >> 4) << 3;
  for (int t = wave; t < 16; t += 8) {
    int m0 = (t >> 2) << 4, n0 = (t & 3) << 4;
    v8f c = {0.f, 0.f, 0.f, 0.f, 0.f, 0.f, 0.f, 0.f};
    for (int k0 = 0; k0 < 64; k0 += 32) {
      v16h a = load_a_lds(P, TST, m0, k0, lane);
      v16h b = load_b_lds(V, TST, k0, n0, lane);
      c = wmma_f16(a, b, c);
    }
    for (int r = 0; r < 8; ++r) {
      int m = m0 + r + mo, n = n0 + e;
      out[m * TST + n] = (h16)(c[r] + pos[m * 64 + n]);
    }
  }
}

// O(64x64 f32) = P @ V, scattered to global: gout[base + m*rowstride + n]
__device__ __forceinline__ void gemm_pv_glb(const h16* P, const h16* V,
                                            float* gout, long rowstride,
                                            long base, int tid) {
  int lane = tid & 31, wave = tid >> 5;
  int e = lane & 15, mo = (lane >> 4) << 3;
  for (int t = wave; t < 16; t += 8) {
    int m0 = (t >> 2) << 4, n0 = (t & 3) << 4;
    v8f c = {0.f, 0.f, 0.f, 0.f, 0.f, 0.f, 0.f, 0.f};
    for (int k0 = 0; k0 < 64; k0 += 32) {
      v16h a = load_a_lds(P, TST, m0, k0, lane);
      v16h b = load_b_lds(V, TST, k0, n0, lane);
      c = wmma_f16(a, b, c);
    }
    for (int r = 0; r < 8; ++r) {
      long m = m0 + r + mo, n = n0 + e;
      gout[base + m * rowstride + n] = c[r];
    }
  }
}

// Parallel row softmax over 64x64 f32 scores (scale = d^-0.5 = 0.125).
// All 256 threads: 4 lanes per row, 16 register-resident elements each;
// max/sum reduced with __shfl_xor(1,2) inside the wave32 quad.  f16 P out.
__device__ __forceinline__ void softmax64(const float* Sf, h16* P, int tid) {
  int row = tid >> 2;
  int sub = tid & 3;
  const float* rp = Sf + row * SST + sub * 16;
  float v[16];
  float mx = -1e30f;
#pragma unroll
  for (int j = 0; j < 16; ++j) {
    v[j] = rp[j] * 0.125f;
    mx = fmaxf(mx, v[j]);
  }
  mx = fmaxf(mx, __shfl_xor(mx, 1));
  mx = fmaxf(mx, __shfl_xor(mx, 2));
  float sum = 0.f;
#pragma unroll
  for (int j = 0; j < 16; ++j) {
    v[j] = __expf(v[j] - mx);
    sum += v[j];
  }
  sum += __shfl_xor(sum, 1);
  sum += __shfl_xor(sum, 2);
  float inv = 1.f / sum;
  h16* pr = P + row * TST + sub * 16;
#pragma unroll
  for (int j = 0; j < 16; ++j) pr[j] = (h16)(v[j] * inv);
}

// ---- kernels ---------------------------------------------------------------

// Pre-transpose + f16-convert weights: wT[k*ldo + n] = w[n*ldi + k]
__global__ __launch_bounds__(256) void prep_weights(
    const float* __restrict__ w1, const float* __restrict__ w2,
    const float* __restrict__ w3, const float* __restrict__ w4,
    h16* __restrict__ w1T, h16* __restrict__ w2T,
    h16* __restrict__ w3T, h16* __restrict__ w4T) {
  int idx = blockIdx.x * blockDim.x + threadIdx.x;
  if (idx < 512 * 1536) {
    int k = idx / 1536, n = idx % 1536;
    w1T[idx] = (h16)w1[n * 512 + k];
    w2T[idx] = (h16)w2[n * 512 + k];
  }
  if (idx < 64 * 192) {
    int k = idx / 192, n = idx % 192;
    w3T[idx] = (h16)w3[n * 64 + k];
    w4T[idx] = (h16)w4[n * 64 + k];
  }
}

__global__ __launch_bounds__(256) void axial_kernel(
    const float* __restrict__ x,
    const float* __restrict__ n3w, const float* __restrict__ n3b,
    const float* __restrict__ n4w, const float* __restrict__ n4b,
    const h16* __restrict__ w1T, const h16* __restrict__ w2T,
    const h16* __restrict__ w3T, const h16* __restrict__ w4T,
    const float* __restrict__ pos1, const float* __restrict__ pos2,
    const float* __restrict__ pos3, const float* __restrict__ pos4,
    float* __restrict__ o1buf, float* __restrict__ o2buf) {
  extern __shared__ char smem[];
  h16* Aw  = (h16*)smem;            // 64 x AST (LN'd window rows,  f16)
  h16* Ah  = Aw + 64 * AST;         // 64 x AST (LN'd column rows, f16)
  h16* Wq  = Ah + 64 * AST;         // per-head q tile / P tile
  h16* WkT = Wq + TBUF;             // per-head k^T tile
  h16* Wv  = WkT + TBUF;            // per-head v tile
  h16* Wx2 = Wv + TBUF;             // attn out + pos
  h16* sQ2[2]  = { Wx2 + 1 * TBUF, Wx2 + 2 * TBUF };
  h16* sK2T[2] = { Wx2 + 3 * TBUF, Wx2 + 4 * TBUF };
  h16* sV2[2]  = { Wx2 + 5 * TBUF, Wx2 + 6 * TBUF };
  float* Sf = (float*)(Wx2 + 7 * TBUF);   // 64 x SST f32 scores

  int tid = threadIdx.x;
  int wg = blockIdx.x;
  int b = wg >> 6;    // batch
  int i = wg & 63;    // grid line index

  // ---- LayerNorm + pos for both branches (4 lanes per row) ----
  {
    int row = tid >> 2;
    int q4 = tid & 3;
    int e0 = q4 * 128;
    // window branch: token (i, row)
    const float* xr = x + (((long)b * 4096) + i * 64 + row) * 512;
    float s = 0.f, s2 = 0.f;
    for (int e = e0; e < e0 + 128; ++e) { float v = xr[e]; s += v; s2 += v * v; }
    s  += __shfl_xor(s, 1);  s  += __shfl_xor(s, 2);
    s2 += __shfl_xor(s2, 1); s2 += __shfl_xor(s2, 2);
    float mu = s * (1.0f / 512.0f);
    float rs = rsqrtf(s2 * (1.0f / 512.0f) - mu * mu + 1e-5f);
    for (int e = e0; e < e0 + 128; ++e)
      Aw[row * AST + e] =
          (h16)((xr[e] - mu) * rs * n3w[e] + n3b[e] + pos1[row * 512 + e]);
    // height branch: token (row, i)
    const float* xc = x + (((long)b * 4096) + row * 64 + i) * 512;
    s = 0.f; s2 = 0.f;
    for (int e = e0; e < e0 + 128; ++e) { float v = xc[e]; s += v; s2 += v * v; }
    s  += __shfl_xor(s, 1);  s  += __shfl_xor(s, 2);
    s2 += __shfl_xor(s2, 1); s2 += __shfl_xor(s2, 2);
    mu = s * (1.0f / 512.0f);
    rs = rsqrtf(s2 * (1.0f / 512.0f) - mu * mu + 1e-5f);
    for (int e = e0; e < e0 + 128; ++e)
      Ah[row * AST + e] =
          (h16)((xc[e] - mu) * rs * n4w[e] + n4b[e] + pos2[row * 512 + e]);
  }
  __syncthreads();

  // ---- per-head pipeline (heads fully independent) ----
  for (int h = 0; h < 8; ++h) {
    for (int br = 0; br < 2; ++br) {
      const h16* A       = br ? Ah : Aw;
      const h16* wqkvT   = br ? w2T : w1T;
      const float* posb  = (br ? pos4 : pos3) + h * 4096;  // (H,64,64)
      const h16* whT     = br ? w4T : w3T;
      // QKV slices for head h: cols h*64 / 512+h*64 / 1024+h*64 of (512x1536)
      gemm_qkv(A, AST, wqkvT, 1536, h * 64,        512, Wq,  false, tid);
      gemm_qkv(A, AST, wqkvT, 1536, 512 + h * 64,  512, WkT, true,  tid);
      gemm_qkv(A, AST, wqkvT, 1536, 1024 + h * 64, 512, Wv,  false, tid);
      __syncthreads();
      // self attention
      gemm_scores(Wq, WkT, Sf, tid);
      __syncthreads();
      softmax64(Sf, Wq, tid);               // P into Wq
      __syncthreads();
      gemm_pv_x2(Wq, Wv, posb, Wx2, tid);   // X2 = P@V + pos
      __syncthreads();
      // head re-projection (64x64 @ 64x192): q2 | k2 | v2
      gemm_qkv(Wx2, TST, whT, 192, 0,   64, sQ2[br],  false, tid);
      gemm_qkv(Wx2, TST, whT, 192, 64,  64, sK2T[br], true,  tid);
      gemm_qkv(Wx2, TST, whT, 192, 128, 64, sV2[br],  false, tid);
      __syncthreads();
    }
    // o1 = attn(xh_q2, xw_k2, xw_v2) -> out token (j, i)
    gemm_scores(sQ2[1], sK2T[0], Sf, tid);
    __syncthreads();
    softmax64(Sf, Wq, tid);
    __syncthreads();
    gemm_pv_glb(Wq, sV2[0], o1buf, 64 * 512,
                (((long)b * 4096) + i) * 512 + h * 64, tid);
    __syncthreads();
    // o2 = attn(xw_q2, xh_k2, xh_v2) -> out token (i, j)
    gemm_scores(sQ2[0], sK2T[1], Sf, tid);
    __syncthreads();
    softmax64(Sf, Wq, tid);
    __syncthreads();
    gemm_pv_glb(Wq, sV2[1], o2buf, 512,
                (((long)b * 4096) + i * 64) * 512 + h * 64, tid);
    __syncthreads();
  }
}

__global__ __launch_bounds__(256) void finalize_kernel(
    const float* __restrict__ x, const float* __restrict__ o1,
    const float* __restrict__ o2, float* __restrict__ out) {
  long idx = ((long)blockIdx.x * blockDim.x + threadIdx.x) * 4;
  float4 a = *(const float4*)(x + idx);
  float4 p = *(const float4*)(o1 + idx);
  float4 q = *(const float4*)(o2 + idx);
  float4 r;
  r.x = a.x + p.x + q.x;
  r.y = a.y + p.y + q.y;
  r.z = a.z + p.z + q.z;
  r.w = a.w + p.w + q.w;
  *(float4*)(out + idx) = r;
}

// ---- launch ----------------------------------------------------------------

extern "C" void kernel_launch(void* const* d_in, const int* in_sizes, int n_in,
                              void* d_out, int out_size, void* d_ws,
                              size_t ws_size, hipStream_t stream) {
  (void)in_sizes; (void)n_in; (void)out_size; (void)ws_size;
  const float* x    = (const float*)d_in[0];
  const float* n3w  = (const float*)d_in[1];
  const float* n3b  = (const float*)d_in[2];
  const float* n4w  = (const float*)d_in[3];
  const float* n4b  = (const float*)d_in[4];
  const float* ln1w = (const float*)d_in[5];
  const float* ln2w = (const float*)d_in[6];
  const float* ln3w = (const float*)d_in[7];
  const float* ln4w = (const float*)d_in[8];
  const float* pos1 = (const float*)d_in[9];
  const float* pos2 = (const float*)d_in[10];
  const float* pos3 = (const float*)d_in[11];
  const float* pos4 = (const float*)d_in[12];

  // workspace layout
  h16* w1T = (h16*)d_ws;                 // 512*1536 f16
  h16* w2T = w1T + 512 * 1536;
  h16* w3T = w2T + 512 * 1536;           // 64*192 f16
  h16* w4T = w3T + 64 * 192;
  float* o1buf = (float*)((char*)d_ws + (4l << 20));   // 4MB-aligned staging
  float* o2buf = o1buf + (long)16 * 4096 * 512;

  prep_weights<<<(512 * 1536 + 255) / 256, 256, 0, stream>>>(
      ln1w, ln2w, ln3w, ln4w, w1T, w2T, w3T, w4T);

  constexpr int SMEM_BYTES = (2 * 64 * AST + 10 * TBUF) * 2 + 64 * SST * 4;
  axial_kernel<<<1024, 256, SMEM_BYTES, stream>>>(
      x, n3w, n3b, n4w, n4b, w1T, w2T, w3T, w4T,
      pos1, pos2, pos3, pos4, o1buf, o2buf);

  finalize_kernel<<<(16 * 4096 * 512 / 4) / 256, 256, 0, stream>>>(
      x, o1buf, o2buf, (float*)d_out);
}
